// MultiHead2_16569983828125
// MI455X (gfx1250) — matrix operand order repaired and checked by reading, not compiled, not run on hardware
//
#include <hip/hip_runtime.h>
#include <hip/hip_bf16.h>

#define BATCH  4
#define SEQ    2048
#define DMODEL 1024
#define NH     16
#define DH     64

typedef __bf16 bf16_t;
typedef __attribute__((ext_vector_type(16))) __bf16 v16bf;
typedef __attribute__((ext_vector_type(8)))  __bf16 v8bf;
typedef __attribute__((ext_vector_type(8)))  float  v8f;
typedef __attribute__((ext_vector_type(4)))  float  v4f;

// ---------------------------------------------------------------------------
// WMMA fragment layout used here (v_wmma_f32_16x16x32_bf16, wave32; own k order, same for A and B)
//
// A (16x32 bf16): lane l -> row = l&15, koff = (l>=16)?16:0
//   elems 0..15 = K[koff+0 .. koff+15]   (contiguous along K)
// B (32x16 bf16): lane l -> col = l&15, koff16 = (l>=16)?16:0
//   elems 0..15 = K[koff16+0 .. koff16+15]   (contiguous along K)
// C/D (16x16 f32): lane l -> col = l&15, elem i -> row = ((l>=16)?8:0)+i
// ---------------------------------------------------------------------------

static __device__ __forceinline__ v8f wmma_bf16(v16bf a, v16bf b, v8f c) {
  return __builtin_amdgcn_wmma_f32_16x16x32_bf16(
      /*neg_a=*/false, a, /*neg_b=*/false, b,
      /*c_mod=*/(short)0, c, /*reuse_a=*/false, /*reuse_b=*/false);
}

static __device__ __forceinline__ v16bf cat8(v8bf lo, v8bf hi) {
  v16bf r;
#pragma unroll
  for (int i = 0; i < 8; ++i) { r[i] = lo[i]; r[8 + i] = hi[i]; }
  return r;
}

// A fragment from bf16 row-major source (global or LDS)
static __device__ __forceinline__ v16bf a_frag_bf16(const bf16_t* row, int koff) {
  v8bf x0 = *(const v8bf*)(row + koff);
  v8bf x1 = *(const v8bf*)(row + koff + 8);
  return cat8(x0, x1);
}

// Reductions across a 16-lane half of the wave (rows of a C fragment)
static __device__ __forceinline__ float half_max(float v) {
#pragma unroll
  for (int m = 8; m >= 1; m >>= 1) v = fmaxf(v, __shfl_xor(v, m, 32));
  return v;
}
static __device__ __forceinline__ float half_sum(float v) {
#pragma unroll
  for (int m = 8; m >= 1; m >>= 1) v += __shfl_xor(v, m, 32);
  return v;
}

// ---------------------------------------------------------------------------
// fp32 -> bf16 bulk conversion (vectorized: 8 elems / thread)
// ---------------------------------------------------------------------------
__global__ __launch_bounds__(256) void cvt_f32_bf16_kernel(
    const float* __restrict__ in, bf16_t* __restrict__ out, int n8) {
  const int t = blockIdx.x * 256 + threadIdx.x;
  if (t >= n8) return;
  const v4f* p = (const v4f*)(in + (size_t)t * 8);
  v4f x0 = p[0], x1 = p[1];
  v8bf y;
#pragma unroll
  for (int i = 0; i < 4; ++i) {
    y[i]     = (__bf16)x0[i];
    y[4 + i] = (__bf16)x1[i];
  }
  *(v8bf*)(out + (size_t)t * 8) = y;
}

// ---------------------------------------------------------------------------
// Q/K/V projection: out = X @ W^T + bias (all bf16 operands).
// transpose_out==0: out[b][h][t][d] ; transpose_out==1: out[b][h][d][t]
// Wave = 32x64 output tile (2 A-frags x 4 B-frags, 8 accumulators);
// block = 8 waves = 256 rows x 64 cols.
// ---------------------------------------------------------------------------
__global__ __launch_bounds__(256) void proj_bf16_kernel(
    const bf16_t* __restrict__ X, const bf16_t* __restrict__ W,
    const float* __restrict__ bias, bf16_t* __restrict__ out,
    int transpose_out) {
  const int lane = threadIdx.x & 31;
  const int wave = threadIdx.x >> 5;
  const int l15 = lane & 15;
  const int hi = lane >> 4;
  const int m0 = blockIdx.x * 256 + wave * 32;
  const int n0 = blockIdx.y * 64;
  const int akoff = hi * 16;
  const int bkoff = hi * 16;

  v8f zero8 = {0, 0, 0, 0, 0, 0, 0, 0};
  v8f acc0[4] = {zero8, zero8, zero8, zero8};
  v8f acc1[4] = {zero8, zero8, zero8, zero8};

  const bf16_t* aprow0 = X + (size_t)(m0 + l15) * DMODEL;
  const bf16_t* aprow1 = aprow0 + (size_t)16 * DMODEL;

  for (int kk = 0; kk < DMODEL; kk += 32) {
    // ---- hoist ALL raw loads of this k-step (independent b128s) ----
    v8bf a0l = *(const v8bf*)(aprow0 + kk + akoff);
    v8bf a0h = *(const v8bf*)(aprow0 + kk + 8 + akoff);
    v8bf a1l = *(const v8bf*)(aprow1 + kk + akoff);
    v8bf a1h = *(const v8bf*)(aprow1 + kk + 8 + akoff);
    v8bf br[8];
#pragma unroll
    for (int j = 0; j < 4; ++j) {
      const bf16_t* wp = W + (size_t)(n0 + j * 16 + l15) * DMODEL + kk + bkoff;
      br[2 * j]     = *(const v8bf*)(wp);
      br[2 * j + 1] = *(const v8bf*)(wp + 8);
    }
    // ---- 8 back-to-back WMMAs over distinct accumulators ----
    const v16bf a0 = cat8(a0l, a0h);
    const v16bf a1 = cat8(a1l, a1h);
#pragma unroll
    for (int j = 0; j < 4; ++j) {
      const v16bf bf = cat8(br[2 * j], br[2 * j + 1]);
      acc0[j] = wmma_bf16(a0, bf, acc0[j]);
      acc1[j] = wmma_bf16(a1, bf, acc1[j]);
    }
  }

  const int moff = hi * 8;
#pragma unroll
  for (int j = 0; j < 4; ++j) {
    const int n = n0 + j * 16 + l15;
    const float bv = bias[n];
    const int h = n >> 6, d = n & (DH - 1);
#pragma unroll
    for (int g = 0; g < 2; ++g) {
      const v8f* acc = g ? &acc1[j] : &acc0[j];
#pragma unroll
      for (int i = 0; i < 8; ++i) {
        const int m = m0 + g * 16 + moff + i;
        const int b = m >> 11, t = m & (SEQ - 1);
        const float val = (*acc)[i] + bv;
        const size_t idx = transpose_out
            ? ((size_t)((b * NH + h) * DH + d)) * SEQ + t
            : ((size_t)((b * NH + h) * SEQ + t)) * DH + d;
        out[idx] = (bf16_t)val;
      }
    }
  }
}

// ---------------------------------------------------------------------------
// Flash attention (causal + key padding mask), bf16 WMMA, fp32 softmax.
// grid.x = B*NH, grid.y = SEQ/128; wave handles 16 query rows.
// ---------------------------------------------------------------------------
__global__ __launch_bounds__(256) void flash_attn_kernel(
    const bf16_t* __restrict__ Q,   // [B,H,T,DH]
    const bf16_t* __restrict__ K,   // [B,H,T,DH]
    const bf16_t* __restrict__ Vt,  // [B,H,DH,T]
    const float* __restrict__ mask, // [B,T]
    bf16_t* __restrict__ O) {       // [B,T,D] head-concatenated
  __shared__ __align__(16) bf16_t pbuf[8][16][64];

  const int lane = threadIdx.x & 31;
  const int wave = threadIdx.x >> 5;
  const int l15 = lane & 15;
  const int hi = lane >> 4;
  const int moff = hi * 8;
  const int akoff = hi * 16;
  const int bkoff = hi * 16;
  const int bh = blockIdx.x;
  const int b = bh >> 4, h = bh & 15;
  const int q0 = blockIdx.y * 128 + wave * 16;

  const bf16_t* Qbh = Q + (size_t)bh * SEQ * DH;
  const bf16_t* Kbh = K + (size_t)bh * SEQ * DH;
  const bf16_t* Vbh = Vt + (size_t)bh * DH * SEQ;
  const float* mrow = mask + (size_t)b * SEQ;

  // Q strip held in registers: two 16x32 A fragments (dh 0..31, 32..63)
  const v16bf qa0 = a_frag_bf16(Qbh + (size_t)(q0 + l15) * DH, akoff);
  const v16bf qa1 = a_frag_bf16(Qbh + (size_t)(q0 + l15) * DH + 32, akoff);

  v8f zero8 = {0, 0, 0, 0, 0, 0, 0, 0};
  v8f o[4] = {zero8, zero8, zero8, zero8};
  float rm[8], rl[8];
#pragma unroll
  for (int i = 0; i < 8; ++i) { rm[i] = -3.0e38f; rl[i] = 0.0f; }

  const float inv_s = 1.0f / 32.0f;  // 1/sqrt(D), D = 1024
  const int kend = q0 + 16;          // causal: keys beyond strip not needed

  for (int kb = 0; kb < kend; kb += 64) {
    // ---- S = Q K^T for a 16x64 block (4 tiles of 16x16) ----
    v8f s[4];
#pragma unroll
    for (int n16 = 0; n16 < 4; ++n16) {
      const bf16_t* kp = Kbh + (size_t)(kb + n16 * 16 + l15) * DH;
      // hoist 4 independent b128 loads, then 2 WMMAs
      v8bf k0l = *(const v8bf*)(kp + bkoff);
      v8bf k0h = *(const v8bf*)(kp + bkoff + 8);
      v8bf k1l = *(const v8bf*)(kp + 32 + bkoff);
      v8bf k1h = *(const v8bf*)(kp + 32 + bkoff + 8);
      v8f sa = zero8;
      sa = wmma_bf16(qa0, cat8(k0l, k0h), sa);
      sa = wmma_bf16(qa1, cat8(k1l, k1h), sa);
      s[n16] = sa;
    }
    // ---- scale + causal + padding mask ----
#pragma unroll
    for (int n16 = 0; n16 < 4; ++n16) {
      const int col = kb + n16 * 16 + l15;
      const float mv = mrow[col];
#pragma unroll
      for (int i = 0; i < 8; ++i) {
        const int row = q0 + moff + i;
        float v = s[n16][i];
        if (col > row) v -= 1.0e10f;       // causal
        v -= (1.0f - mv) * 1.0e10f;        // key padding
        s[n16][i] = v * inv_s;
      }
    }
    // ---- online softmax: row max, rescale ----
    float nm[8], sc[8];
#pragma unroll
    for (int i = 0; i < 8; ++i) {
      float m4 = fmaxf(fmaxf(s[0][i], s[1][i]), fmaxf(s[2][i], s[3][i]));
      m4 = half_max(m4);
      const float newm = fmaxf(rm[i], m4);
      sc[i] = __expf(rm[i] - newm);
      nm[i] = newm;
      rm[i] = newm;
    }
#pragma unroll
    for (int j = 0; j < 4; ++j)
#pragma unroll
      for (int i = 0; i < 8; ++i) o[j][i] *= sc[i];
    // ---- P = exp(S - m), row sums; stage P to LDS in row-major ----
#pragma unroll
    for (int i = 0; i < 8; ++i) {
      float rs = 0.0f;
#pragma unroll
      for (int n16 = 0; n16 < 4; ++n16) {
        const float p = __expf(s[n16][i] - nm[i]);
        rs += half_sum(p);
        pbuf[wave][moff + i][n16 * 16 + l15] = (bf16_t)p;
      }
      rl[i] = rl[i] * sc[i] + rs;
    }
    asm volatile("s_wait_dscnt 0" ::: "memory");  // per-wave LDS RAW fence
    // ---- reload P as A fragments (layout shuffle via LDS) ----
    const v16bf pa0 = a_frag_bf16(&pbuf[wave][l15][0], akoff);
    const v16bf pa1 = a_frag_bf16(&pbuf[wave][l15][32], akoff);
    // ---- O += P @ V (V pre-transposed: contiguous along keys) ----
#pragma unroll
    for (int j = 0; j < 4; ++j) {
      const bf16_t* vp = Vbh + (size_t)(j * 16 + l15) * SEQ + kb;
      v8bf v0l = *(const v8bf*)(vp + bkoff);
      v8bf v0h = *(const v8bf*)(vp + bkoff + 8);
      v8bf v1l = *(const v8bf*)(vp + 32 + bkoff);
      v8bf v1h = *(const v8bf*)(vp + 32 + bkoff + 8);
      o[j] = wmma_bf16(pa0, cat8(v0l, v0h), o[j]);
      o[j] = wmma_bf16(pa1, cat8(v1l, v1h), o[j]);
    }
  }

  // ---- normalize and emit head-concatenated bf16 ----
#pragma unroll
  for (int i = 0; i < 8; ++i) {
    const float inv = 1.0f / fmaxf(rl[i], 1e-30f);
    const int row = q0 + moff + i;
    bf16_t* orow = O + ((size_t)(b * SEQ + row)) * DMODEL + h * DH;
#pragma unroll
    for (int j = 0; j < 4; ++j) orow[j * 16 + l15] = (bf16_t)(o[j][i] * inv);
  }
}

// ---------------------------------------------------------------------------
// Output projection: d_out = attn(bf16) @ Wo^T + bo, fp32 result.
// Same 32x64 wave tile as proj_bf16_kernel.
// ---------------------------------------------------------------------------
__global__ __launch_bounds__(256) void out_proj_kernel(
    const bf16_t* __restrict__ A, const bf16_t* __restrict__ W,
    const float* __restrict__ bias, float* __restrict__ Cout) {
  const int lane = threadIdx.x & 31;
  const int wave = threadIdx.x >> 5;
  const int l15 = lane & 15;
  const int hi = lane >> 4;
  const int m0 = blockIdx.x * 256 + wave * 32;
  const int n0 = blockIdx.y * 64;
  const int akoff = hi * 16;
  const int bkoff = hi * 16;

  v8f zero8 = {0, 0, 0, 0, 0, 0, 0, 0};
  v8f acc0[4] = {zero8, zero8, zero8, zero8};
  v8f acc1[4] = {zero8, zero8, zero8, zero8};

  const bf16_t* aprow0 = A + (size_t)(m0 + l15) * DMODEL;
  const bf16_t* aprow1 = aprow0 + (size_t)16 * DMODEL;

  for (int kk = 0; kk < DMODEL; kk += 32) {
    v8bf a0l = *(const v8bf*)(aprow0 + kk + akoff);
    v8bf a0h = *(const v8bf*)(aprow0 + kk + 8 + akoff);
    v8bf a1l = *(const v8bf*)(aprow1 + kk + akoff);
    v8bf a1h = *(const v8bf*)(aprow1 + kk + 8 + akoff);
    v8bf br[8];
#pragma unroll
    for (int j = 0; j < 4; ++j) {
      const bf16_t* wp = W + (size_t)(n0 + j * 16 + l15) * DMODEL + kk + bkoff;
      br[2 * j]     = *(const v8bf*)(wp);
      br[2 * j + 1] = *(const v8bf*)(wp + 8);
    }
    const v16bf a0 = cat8(a0l, a0h);
    const v16bf a1 = cat8(a1l, a1h);
#pragma unroll
    for (int j = 0; j < 4; ++j) {
      const v16bf bf = cat8(br[2 * j], br[2 * j + 1]);
      acc0[j] = wmma_bf16(a0, bf, acc0[j]);
      acc1[j] = wmma_bf16(a1, bf, acc1[j]);
    }
  }

  const int moff = hi * 8;
#pragma unroll
  for (int j = 0; j < 4; ++j) {
    const int n = n0 + j * 16 + l15;
    const float bv = bias[n];
#pragma unroll
    for (int i = 0; i < 8; ++i) {
      Cout[(size_t)(m0 + moff + i) * DMODEL + n] = acc0[j][i] + bv;
      Cout[(size_t)(m0 + 16 + moff + i) * DMODEL + n] = acc1[j][i] + bv;
    }
  }
}

// ---------------------------------------------------------------------------
extern "C" void kernel_launch(void* const* d_in, const int* in_sizes, int n_in,
                              void* d_out, int out_size, void* d_ws,
                              size_t ws_size, hipStream_t stream) {
  (void)in_sizes; (void)n_in; (void)out_size; (void)ws_size;
  const float* query = (const float*)d_in[0];
  const float* key   = (const float*)d_in[1];
  const float* value = (const float*)d_in[2];
  const float* mask  = (const float*)d_in[3];
  const float* Wq = (const float*)d_in[4];
  const float* bq = (const float*)d_in[5];
  const float* Wk = (const float*)d_in[6];
  const float* bk = (const float*)d_in[7];
  const float* Wv = (const float*)d_in[8];
  const float* bv = (const float*)d_in[9];
  const float* Wo = (const float*)d_in[10];
  const float* bo = (const float*)d_in[11];

  // Workspace layout (all bf16):
  //   qb, kb, vt, attn : B*T*D each (16.78 MB)     -> 67.1 MB
  //   xq, xk, xv       : B*T*D each (bf16 inputs)  -> 50.3 MB
  //   wqb,wkb,wvb,wob  : D*D each (bf16 weights)   ->  8.4 MB
  const size_t elems = (size_t)BATCH * SEQ * DMODEL;  // 8388608
  const size_t welems = (size_t)DMODEL * DMODEL;      // 1048576
  bf16_t* qb   = (bf16_t*)d_ws;
  bf16_t* kb   = qb + elems;
  bf16_t* vt   = kb + elems;   // V stored transposed [b,h,dh,t]
  bf16_t* attn = vt + elems;
  bf16_t* xq   = attn + elems;
  bf16_t* xk   = xq + elems;
  bf16_t* xv   = xk + elems;
  bf16_t* wqb  = xv + elems;
  bf16_t* wkb  = wqb + welems;
  bf16_t* wvb  = wkb + welems;
  bf16_t* wob  = wvb + welems;

  dim3 blk(256);
  // --- one-time fp32 -> bf16 conversions ---
  const int act8 = (int)(elems / 8), w8 = (int)(welems / 8);
  cvt_f32_bf16_kernel<<<dim3((act8 + 255) / 256), blk, 0, stream>>>(query, xq, act8);
  cvt_f32_bf16_kernel<<<dim3((act8 + 255) / 256), blk, 0, stream>>>(key,   xk, act8);
  cvt_f32_bf16_kernel<<<dim3((act8 + 255) / 256), blk, 0, stream>>>(value, xv, act8);
  cvt_f32_bf16_kernel<<<dim3((w8 + 255) / 256),   blk, 0, stream>>>(Wq, wqb, w8);
  cvt_f32_bf16_kernel<<<dim3((w8 + 255) / 256),   blk, 0, stream>>>(Wk, wkb, w8);
  cvt_f32_bf16_kernel<<<dim3((w8 + 255) / 256),   blk, 0, stream>>>(Wv, wvb, w8);
  cvt_f32_bf16_kernel<<<dim3((w8 + 255) / 256),   blk, 0, stream>>>(Wo, wob, w8);

  // --- projections (pure bf16 GEMMs) ---
  dim3 gproj(BATCH * SEQ / 256, DMODEL / 64);  // (32, 16)
  proj_bf16_kernel<<<gproj, blk, 0, stream>>>(xq, wqb, bq, qb, 0);
  proj_bf16_kernel<<<gproj, blk, 0, stream>>>(xk, wkb, bk, kb, 0);
  proj_bf16_kernel<<<gproj, blk, 0, stream>>>(xv, wvb, bv, vt, 1);

  // --- flash attention ---
  dim3 gattn(BATCH * NH, SEQ / 128);           // (64, 16)
  flash_attn_kernel<<<gattn, blk, 0, stream>>>(qb, kb, vt, mask, attn);

  // --- output projection (fp32 result) ---
  out_proj_kernel<<<gproj, blk, 0, stream>>>(attn, wob, bo, (float*)d_out);
}
